// FrequencyDynamicSelection_67843303408303
// MI455X (gfx1250) — matrix operand order repaired and checked by reading, not compiled, not run on hardware
//
#include <hip/hip_runtime.h>
#include <hip/hip_bf16.h>

// ---------------------------------------------------------------------------
// FrequencyDynamicSelection for MI455X (gfx1250, wave32, WMMA)
//
// reference = circular depthwise 7x7 conv with a per-(b,c) dynamic kernel:
//   out[b,c,y,x] = sum_{i,j} dyn[b,c,i,j] * x[b,c,(y+3-i)%H,(x+3-j)%W]
//   dyn[b,c]     = sum_k softmax(MLP(meanpool(guidance[b])))[k] * basis[k,c]
//
// Conv mapped to v_wmma_f32_16x16x32_f16 via banded Toeplitz:
//   per 16x16 output tile: Out = sum_{i=0..6} A_i(16x32,f16) @ T_i(32x16,f16)
//   A_i[m,k] = x[(y0+m+3-i)%H, (x0+k-3)%W]   (K padded 22->32)
//   T_i[k,n] = dyn[i, n+6-k] if 0<=n+6-k<7 else 0   (zero rows kill K>=22 pad)
// ---------------------------------------------------------------------------

#define B_  8
#define C_  128
#define H_  256
#define W_  256
#define K_  4
#define KS_ 7

typedef __attribute__((ext_vector_type(16))) _Float16 v16h;
typedef __attribute__((ext_vector_type(8)))  float    v8f;

union V16H { unsigned u[8]; v16h v; };

static __device__ inline unsigned short f2h_bits(float f) {
    _Float16 h = (_Float16)f;                 // v_cvt_f16_f32, RTE
    union { _Float16 h; unsigned short s; } c; c.h = h; return c.s;
}
static __device__ inline unsigned pack2(float lo, float hi) {
    return (unsigned)f2h_bits(lo) | ((unsigned)f2h_bits(hi) << 16);
}

// ---------------- Kernel 1: mean pool guidance -> g[B][C] -------------------
__global__ __launch_bounds__(256)
void fds_pool(const float* __restrict__ gsrc, float* __restrict__ g) {
    __shared__ float red[256];
    const int bc = blockIdx.x;                       // b*C + c
    const float4* p = (const float4*)(gsrc + (size_t)bc * (H_ * W_));
    const int t = threadIdx.x;
    float s = 0.f;
#pragma unroll 4
    for (int it = 0; it < (H_ * W_ / 4) / 256; ++it) {
        float4 v = p[t + it * 256];
        s += (v.x + v.y) + (v.z + v.w);
    }
    red[t] = s;
    __syncthreads();
    for (int off = 128; off > 0; off >>= 1) {
        if (t < off) red[t] += red[t + off];
        __syncthreads();
    }
    if (t == 0) g[bc] = red[0] * (1.0f / (H_ * W_));
}

// ---------------- Kernel 2: MLP + softmax -> weights[B][K] ------------------
__global__ __launch_bounds__(256)
void fds_mlp(const float* __restrict__ g,
             const float* __restrict__ w1, const float* __restrict__ b1,
             const float* __restrict__ w2, const float* __restrict__ b2,
             float* __restrict__ weights) {
    __shared__ float h1[B_][C_ / 4];
    __shared__ float logits[B_][K_];
    const int t = threadIdx.x;                       // 256 threads
    {   // h1 = relu(g @ w1^T + b1): 8x32 dot-128s, one per thread
        const int b = t >> 5, j = t & 31;
        float s = b1[j];
#pragma unroll 8
        for (int c = 0; c < C_; ++c) s += g[b * C_ + c] * w1[j * C_ + c];
        h1[b][j] = fmaxf(s, 0.f);
    }
    __syncthreads();
    if (t < B_ * K_) {                               // logits = h1 @ w2^T + b2
        const int b = t / K_, k = t % K_;
        float s = b2[k];
#pragma unroll
        for (int j = 0; j < C_ / 4; ++j) s += h1[b][j] * w2[k * (C_ / 4) + j];
        logits[b][k] = s;
    }
    __syncthreads();
    if (t < B_) {                                    // softmax over K=4
        float mx = logits[t][0];
#pragma unroll
        for (int k = 1; k < K_; ++k) mx = fmaxf(mx, logits[t][k]);
        float e[K_], sum = 0.f;
#pragma unroll
        for (int k = 0; k < K_; ++k) { e[k] = __expf(logits[t][k] - mx); sum += e[k]; }
        const float inv = 1.0f / sum;
#pragma unroll
        for (int k = 0; k < K_; ++k) weights[t * K_ + k] = e[k] * inv;
    }
}

// ---------------- Kernel 3: WMMA circular depthwise conv --------------------
// grid = B*C blocks, 256 threads (8 wave32). Block owns one (b,c) plane.
// 16 y-strips; per strip: stage 22 x 272 f16 halo'd patch in LDS (circular
// wrap folded in), then each wave computes two 16x16 tiles with 7 WMMAs.
#define PSTRIDE 274          // f16 elems; even (4B-aligned pairs), 137 dwords/row
#define PROWS   22
#define PCOLS   272

__global__ __launch_bounds__(256)
void fds_conv(const float* __restrict__ x,
              const float* __restrict__ basis,     // [K][C][7][7]
              const float* __restrict__ weights,   // [B][K]
              float* __restrict__ out) {
    __shared__ float dyn[KS_ * KS_];
    __shared__ __align__(4) unsigned short patch[PROWS * PSTRIDE];

    const int bc = blockIdx.x;
    const int b  = bc >> 7;          // / C_
    const int ch = bc & (C_ - 1);
    const int tid = threadIdx.x;

    // dyn[7][7] = sum_k weights[b][k] * basis[k][ch]
    if (tid < KS_ * KS_) {
        float s = 0.f;
#pragma unroll
        for (int k = 0; k < K_; ++k)
            s += weights[b * K_ + k] * basis[(size_t)(k * C_ + ch) * (KS_ * KS_) + tid];
        dyn[tid] = s;
    }
    __syncthreads();

    const int lane = tid & 31;
    const int m    = lane & 15;      // M row / N col index
    const int hgrp = lane >> 4;      // lane half-group
    const int wv   = tid >> 5;       // wave id 0..7

    // Build T_i (B-matrix) in VGPRs: 32x16 f16, lanes 0-15 K=0..15,
    // lanes 16-31 K=16..31, VGPR v holds K pair (2v, 2v+1) within the half.
    unsigned Bm[KS_][8];
#pragma unroll
    for (int i = 0; i < KS_; ++i) {
#pragma unroll
        for (int v = 0; v < 8; ++v) {
            const int k0 = 2 * v + 16 * hgrp;
            const int j0 = m + 6 - k0;           // n = m (lane&15)
            const int j1 = j0 - 1;
            const float f0 = (j0 >= 0 && j0 < KS_) ? dyn[i * KS_ + j0] : 0.f;
            const float f1 = (j1 >= 0 && j1 < KS_) ? dyn[i * KS_ + j1] : 0.f;
            Bm[i][v] = pack2(f0, f1);
        }
    }

    const float* __restrict__ xp = x   + (size_t)bc * (H_ * W_);
    float*       __restrict__ op = out + (size_t)bc * (H_ * W_);

    for (int ys = 0; ys < H_ / 16; ++ys) {
        const int y0 = ys * 16;
        // Stage halo'd strip: LDS row r <-> y=(y0-3+r)%H, col t <-> x=(t-3)%W
        for (int e = tid; e < PROWS * PCOLS; e += 256) {
            const int r  = e / PCOLS;
            const int t  = e - r * PCOLS;
            const int gy = (y0 - 3 + r) & (H_ - 1);
            const int gx = (t - 3) & (W_ - 1);
            patch[r * PSTRIDE + t] = f2h_bits(xp[gy * W_ + gx]);
        }
        __syncthreads();

        for (int xt = wv; xt < 16; xt += 8) {    // uniform per wave
            const int x0 = xt * 16;
            v8f acc = {0.f, 0.f, 0.f, 0.f, 0.f, 0.f, 0.f, 0.f};
#pragma unroll
            for (int i = 0; i < KS_; ++i) {
                // A-matrix 16-bit 16x32: lane half-group picks K offset 16*hgrp,
                // VGPR v -> K pair (2v, 2v+1) within the half (matches B)
                const int base = (m + 6 - i) * PSTRIDE + x0;   // even elem offs
                V16H a;
#pragma unroll
                for (int v = 0; v < 8; ++v) {
                    const int kA = 2 * v + 16 * hgrp;
                    a.u[v] = *(const unsigned*)(&patch[base + kA]);
                }
                V16H bm;
#pragma unroll
                for (int v = 0; v < 8; ++v) bm.u[v] = Bm[i][v];
                acc = __builtin_amdgcn_wmma_f32_16x16x32_f16(
                        false, a.v, false, bm.v, (short)0, acc, false, false);
            }
            // C/D layout: VGPR v -> M = v + 8*hgrp, N = lane&15
#pragma unroll
            for (int v = 0; v < 8; ++v) {
                const int y = y0 + v + 8 * hgrp;
                op[y * W_ + x0 + m] = acc[v];
            }
        }
        __syncthreads();
    }
}

// ---------------------------------------------------------------------------
extern "C" void kernel_launch(void* const* d_in, const int* in_sizes, int n_in,
                              void* d_out, int out_size, void* d_ws, size_t ws_size,
                              hipStream_t stream) {
    const float* x        = (const float*)d_in[0];
    const float* guidance = (const float*)d_in[1];
    const float* basis    = (const float*)d_in[2];
    const float* w1       = (const float*)d_in[3];
    const float* b1       = (const float*)d_in[4];
    const float* w2       = (const float*)d_in[5];
    const float* b2       = (const float*)d_in[6];
    float* out = (float*)d_out;

    float* g_ws   = (float*)d_ws;                 // [B][C]   = 1024 f32
    float* wts_ws = g_ws + B_ * C_;               // [B][K]   = 32 f32

    fds_pool<<<B_ * C_, 256, 0, stream>>>(guidance, g_ws);
    fds_mlp <<<1,       256, 0, stream>>>(g_ws, w1, b1, w2, b2, wts_ws);
    fds_conv<<<B_ * C_, 256, 0, stream>>>(x, basis, wts_ws, out);
}
